// MultiHeadAttention_8160437862644
// MI455X (gfx1250) — matrix operand order repaired and checked by reading, not compiled, not run on hardware
//
#include <hip/hip_runtime.h>

// ---------------------------------------------------------------------------
// MHA forward for gfx1250 (MI455X).  f16 WMMA (16x16x32) with f32 accum,
// flash-attention online softmax (no materialized scores), causal predicate
// replaces the tril mask input.  K/V and W tiles are staged into LDS with
// GLOBAL_LOAD_ASYNC_TO_LDS_B128 (ASYNCcnt) and double-buffered so the DMA of
// tile k+1 overlaps the WMMA work on tile k.  Workspace use: 64 MiB of d_ws.
// ---------------------------------------------------------------------------

typedef __attribute__((ext_vector_type(16))) _Float16 v16h;
typedef __attribute__((ext_vector_type(8)))  _Float16 v8h;
typedef __attribute__((ext_vector_type(8)))  float    v8f;

#define BSZ   2
#define SEQ   2048
#define EMBED 1024
#define HEADS 16
#define HDIM  64

__device__ __forceinline__ v8f wmma_f16(v16h a, v16h b, v8f c) {
  // (neg_a, A, neg_b, B, c_mod, C, reuse_a, reuse_b)
  return __builtin_amdgcn_wmma_f32_16x16x32_f16(false, a, false, b, (short)0, c,
                                                false, false);
}

// Async DMA: 16 bytes global -> LDS, tracked by ASYNCcnt.  GV addressing:
// 64-bit global address in a VGPR pair, LDS byte offset in the VDST VGPR
// (generic-pointer low 32 bits == LDS offset per the aperture rules).
__device__ __forceinline__ void async_ld16(const _Float16* gsrc, _Float16* ldst) {
  asm volatile("global_load_async_to_lds_b128 %0, %1, off"
               :: "v"((unsigned int)(uintptr_t)ldst),
                  "v"((unsigned long long)(uintptr_t)gsrc)
               : "memory");
}
__device__ __forceinline__ void wait_async_le4() {
  asm volatile("s_wait_asynccnt 4" ::: "memory");
}
__device__ __forceinline__ void wait_async_le2() {
  asm volatile("s_wait_asynccnt 2" ::: "memory");
}
__device__ __forceinline__ void wait_async_le0() {
  asm volatile("s_wait_asynccnt 0" ::: "memory");
}

// A fragment: 16 rows x 32 K, f16, source row-major with leading dim ld.
// ISA layout: lanes 0-15 -> row=lane, K {0..7,16..23}; lanes 16-31 -> row=lane-16,
// K {8..15,24..31}.  Two 16B loads per lane.
__device__ __forceinline__ v16h load_frag_a(const _Float16* base, int ld) {
  const int lane = threadIdx.x & 31;
  const int row  = lane & 15;
  const int koff = (lane >> 4) << 3;               // 0 or 8
  const _Float16* p = base + row * ld + koff;
  v8h lo = *(const v8h*)(p);
  v8h hi = *(const v8h*)(p + 16);
  v16h r;
#pragma unroll
  for (int i = 0; i < 8; ++i) { r[i] = lo[i]; r[i + 8] = hi[i]; }
  return r;
}

// B fragment (32K x 16N) supplied as B-transposed rows (16 rows x 32 K).
// ISA layout: lane&15 = column, K {0..7,16..23} (lanes 0-15) / {8..15,24..31} (lanes 16-31).
__device__ __forceinline__ v16h load_frag_bt(const _Float16* baseT, int ld) {
  const int lane = threadIdx.x & 31;
  const int row  = lane & 15;
  const int koff = (lane >> 4) << 3;               // 0 or 8
  const _Float16* p = baseT + row * ld + koff;
  v8h lo = *(const v8h*)(p);
  v8h hi = *(const v8h*)(p + 16);
  v16h r;
#pragma unroll
  for (int i = 0; i < 8; ++i) { r[i] = lo[i]; r[i + 8] = hi[i]; }
  return r;
}

// wave32: reduce across the 16 lanes that hold one matrix row's columns.
__device__ __forceinline__ float half16_max(float v) {
  v = fmaxf(v, __shfl_xor(v, 1, 32));
  v = fmaxf(v, __shfl_xor(v, 2, 32));
  v = fmaxf(v, __shfl_xor(v, 4, 32));
  v = fmaxf(v, __shfl_xor(v, 8, 32));
  return v;
}
__device__ __forceinline__ float half16_sum(float v) {
  v += __shfl_xor(v, 1, 32);
  v += __shfl_xor(v, 2, 32);
  v += __shfl_xor(v, 4, 32);
  v += __shfl_xor(v, 8, 32);
  return v;
}

// ---------------------------------------------------------------------------
__global__ void cvt_f32_f16(const float* __restrict__ in,
                            _Float16* __restrict__ out, int n) {
  int i = blockIdx.x * blockDim.x + threadIdx.x;
  const int stride = gridDim.x * blockDim.x;
  for (; i < n; i += stride) out[i] = (_Float16)in[i];
}

// ---------------------------------------------------------------------------
// Y = X[4096,1024] @ W^T + bias, then *scale.
// MODE 0: write f16, rearranged to [b, h, s, d].  MODE 1: write f32 flat [m, f].
// Block: 256 threads (8 waves); tile 128(M) x 128(N); K step 32.
// W tile double-buffered in LDS via async DMA (2 async ops per wave per tile).
template <int MODE>
__global__ __launch_bounds__(256) void proj_gemm(
    const _Float16* __restrict__ X, const _Float16* __restrict__ W,
    const float* __restrict__ bias, _Float16* __restrict__ out16,
    float* __restrict__ out32, float scale) {
  __shared__ alignas(32) _Float16 bT[2][128][32];  // W rows (= B^T), 2 x 8 KiB

  const int w    = threadIdx.x >> 5;
  const int lane = threadIdx.x & 31;
  const int hi   = lane >> 4;
  const int colL = lane & 15;
  const int n0 = blockIdx.x * 128;
  const int mw = blockIdx.y * 128 + w * 16;        // this wave's 16 rows

  auto stage = [&](int k0, int buf) {              // 128 rows x 32 f16 = 8 KiB
    for (int i = threadIdx.x; i < 512; i += 256) {
      int row = i >> 2;
      int cc  = (i & 3) << 3;
      async_ld16(&W[(size_t)(n0 + row) * EMBED + k0 + cc], &bT[buf][row][cc]);
    }
  };

  stage(0, 0);                                     // preload first tile
  v8f acc[8] = {};
  for (int k0 = 0; k0 < EMBED; k0 += 32) {
    const int idx = (k0 >> 5) & 1;
    __syncthreads();                               // readers of buf idx^1 done
    const bool hasNext = (k0 + 32 < EMBED);
    if (hasNext) stage(k0 + 32, idx ^ 1);          // overlap fetch with compute
    if (hasNext) wait_async_le2(); else wait_async_le0();
    __syncthreads();                               // tile k0 visible to all waves
    v16h a = load_frag_a(X + (size_t)mw * EMBED + k0, EMBED);
#pragma unroll
    for (int g = 0; g < 8; ++g)
      acc[g] = wmma_f16(a, load_frag_bt(&bT[idx][g * 16][0], 32), acc[g]);
  }

#pragma unroll
  for (int g = 0; g < 8; ++g) {
    const int f = n0 + g * 16 + colL;
    const float bb = bias[f];
#pragma unroll
    for (int r = 0; r < 8; ++r) {
      const int m = mw + r + hi * 8;               // C layout: M = r (+8 hi half)
      const float val = (acc[g][r] + bb) * scale;
      if (MODE == 0) {
        const int b_ = m >> 11, s_ = m & (SEQ - 1);
        const int h_ = f >> 6,  d_ = f & (HDIM - 1);
        out16[(((size_t)b_ * HEADS + h_) * SEQ + s_) * HDIM + d_] = (_Float16)val;
      } else {
        out32[(size_t)m * EMBED + f] = val;
      }
    }
  }
}

// ---------------------------------------------------------------------------
// Flash attention, causal.  Q pre-scaled by 1/sqrt(HDIM).  Block: 8 waves,
// 128 q-rows; 64-key K/V tiles async-DMAed into double-buffered LDS; V is
// transposed LDS->LDS for the PV B-fragments.
__global__ __launch_bounds__(256) void flash_attn(
    const _Float16* __restrict__ Q, const _Float16* __restrict__ K,
    const _Float16* __restrict__ V, _Float16* __restrict__ O) {
  __shared__ alignas(32) _Float16 kT[2][64][HDIM];   // [k][d]        16 KiB
  __shared__ alignas(32) _Float16 vR[2][64][HDIM];   // raw V [k][d]  16 KiB
  __shared__ alignas(32) _Float16 vT[HDIM][64];      // V^T [d][k]     8 KiB
  __shared__ alignas(32) _Float16 pT[8][16][64];     // per-wave P    16 KiB

  const int w    = threadIdx.x >> 5;
  const int lane = threadIdx.x & 31;
  const int hi   = lane >> 4;
  const int colL = lane & 15;
  const int bh = blockIdx.y;                         // b*16 + h
  const int b_ = bh >> 4, h_ = bh & 15;
  const int qw = blockIdx.x * 128 + w * 16;          // this wave's q rows

  const _Float16* Qb = Q + (size_t)bh * SEQ * HDIM;
  const _Float16* Kb = K + (size_t)bh * SEQ * HDIM;
  const _Float16* Vb = V + (size_t)bh * SEQ * HDIM;

  auto stage = [&](int kbase, int buf) {             // K + V tile, 4 async/wave
    for (int i = threadIdx.x; i < 512; i += 256) {
      int row = i >> 3;
      int cc  = (i & 7) << 3;
      async_ld16(&Kb[(size_t)(kbase + row) * HDIM + cc], &kT[buf][row][cc]);
      async_ld16(&Vb[(size_t)(kbase + row) * HDIM + cc], &vR[buf][row][cc]);
    }
  };

  const v16h aq0 = load_frag_a(Qb + (size_t)qw * HDIM + 0,  HDIM);
  const v16h aq1 = load_frag_a(Qb + (size_t)qw * HDIM + 32, HDIM);

  v8f acc[4] = {};
  float mrun[8], lrun[8];
#pragma unroll
  for (int r = 0; r < 8; ++r) { mrun[r] = -3.0e38f; lrun[r] = 0.0f; }

  const int nkb = (blockIdx.x + 1) * 2;              // causal: keys <= q-tile end
  stage(0, 0);                                       // preload first tile
  for (int kb = 0; kb < nkb; ++kb) {
    const int kbase = kb * 64;
    const int idx = kb & 1;
    const bool hasNext = (kb + 1 < nkb);
    if (hasNext) stage(kbase + 64, idx ^ 1);         // overlap fetch with compute
    if (hasNext) wait_async_le4(); else wait_async_le0();
    __syncthreads();                                 // (A) tile kb in LDS, all waves

    // Transpose V tile LDS->LDS (prior PV readers of vT are behind barrier A).
    for (int i = threadIdx.x; i < 512; i += 256) {
      int row = i >> 3;
      int cc  = (i & 7) << 3;
      v8h vv = *(const v8h*)&vR[idx][row][cc];
#pragma unroll
      for (int j = 0; j < 8; ++j) vT[cc + j][row] = vv[j];
    }

    const bool active = (kbase <= qw + 15);          // else tile fully masked
    if (active) {
      v8f sc[4];
#pragma unroll
      for (int g = 0; g < 4; ++g) {                  // S = Q K^T  (8 WMMA)
        v8f t = {};
        t = wmma_f16(aq0, load_frag_bt(&kT[idx][g * 16][0],  HDIM), t);
        t = wmma_f16(aq1, load_frag_bt(&kT[idx][g * 16][32], HDIM), t);
        sc[g] = t;
      }
      if (kbase + 63 > qw) {                         // causal mask (partial tile)
#pragma unroll
        for (int g = 0; g < 4; ++g) {
          const int n = kbase + g * 16 + colL;
#pragma unroll
          for (int r = 0; r < 8; ++r)
            if (n > qw + r + hi * 8) sc[g][r] = -1.0e30f;
        }
      }
      float fac[8];
#pragma unroll
      for (int r = 0; r < 8; ++r) {                  // online max
        float mx = fmaxf(fmaxf(sc[0][r], sc[1][r]), fmaxf(sc[2][r], sc[3][r]));
        mx = half16_max(mx);
        const float mnew = fmaxf(mrun[r], mx);
        fac[r] = __expf(mrun[r] - mnew);
        mrun[r] = mnew;
      }
#pragma unroll
      for (int g = 0; g < 4; ++g)
#pragma unroll
        for (int r = 0; r < 8; ++r) sc[g][r] = __expf(sc[g][r] - mrun[r]);
#pragma unroll
      for (int r = 0; r < 8; ++r) {                  // online sum
        float sm = sc[0][r] + sc[1][r] + sc[2][r] + sc[3][r];
        lrun[r] = lrun[r] * fac[r] + half16_sum(sm);
      }
#pragma unroll
      for (int g = 0; g < 4; ++g)                    // rescale accumulator
#pragma unroll
        for (int r = 0; r < 8; ++r) acc[g][r] *= fac[r];
#pragma unroll
      for (int g = 0; g < 4; ++g)                    // P -> LDS (A-layout reload)
#pragma unroll
        for (int r = 0; r < 8; ++r)
          pT[w][r + hi * 8][g * 16 + colL] = (_Float16)sc[g][r];
    }
    __syncthreads();                                 // (B) vT & P visible
    if (active) {
      const v16h ap0 = load_frag_a(&pT[w][0][0],  64);
      const v16h ap1 = load_frag_a(&pT[w][0][32], 64);
#pragma unroll
      for (int gd = 0; gd < 4; ++gd) {               // acc += P V  (8 WMMA)
        acc[gd] = wmma_f16(ap0, load_frag_bt(&vT[gd * 16][0],  64), acc[gd]);
        acc[gd] = wmma_f16(ap1, load_frag_bt(&vT[gd * 16][32], 64), acc[gd]);
      }
    }
  }

  float inv[8];
#pragma unroll
  for (int r = 0; r < 8; ++r) inv[r] = 1.0f / lrun[r];   // diag always unmasked
#pragma unroll
  for (int gd = 0; gd < 4; ++gd) {
    const int d = gd * 16 + colL;
#pragma unroll
    for (int r = 0; r < 8; ++r) {
      const int s = qw + r + hi * 8;
      O[((size_t)b_ * SEQ + s) * EMBED + h_ * HDIM + d] =
          (_Float16)(acc[gd][r] * inv[r]);
    }
  }
}

// ---------------------------------------------------------------------------
extern "C" void kernel_launch(void* const* d_in, const int* in_sizes, int n_in,
                              void* d_out, int out_size, void* d_ws,
                              size_t ws_size, hipStream_t stream) {
  (void)in_sizes; (void)n_in; (void)out_size; (void)ws_size;

  const float* q  = (const float*)d_in[0];
  const float* k  = (const float*)d_in[1];
  const float* v  = (const float*)d_in[2];
  // d_in[3] = mask: exactly tril(ones) -> implemented as causal predicate.
  const float* Wq = (const float*)d_in[4];
  const float* bq = (const float*)d_in[5];
  const float* Wk = (const float*)d_in[6];
  const float* bk = (const float*)d_in[7];
  const float* Wv = (const float*)d_in[8];
  const float* bv = (const float*)d_in[9];
  const float* Wo = (const float*)d_in[10];
  const float* bo = (const float*)d_in[11];

  const size_t NX = (size_t)BSZ * SEQ * EMBED;   // 4 Mi elems
  const size_t NW = (size_t)EMBED * EMBED;       // 1 Mi elems

  _Float16* p = (_Float16*)d_ws;                 // 64 MiB total f16 workspace
  _Float16* Xq16 = p; p += NX;
  _Float16* Xk16 = p; p += NX;
  _Float16* Xv16 = p; p += NX;
  _Float16* Wq16 = p; p += NW;
  _Float16* Wk16 = p; p += NW;
  _Float16* Wv16 = p; p += NW;
  _Float16* Wo16 = p; p += NW;
  _Float16* Q16  = p; p += NX;
  _Float16* K16  = p; p += NX;
  _Float16* V16  = p; p += NX;
  _Float16* O16  = p; p += NX;

  cvt_f32_f16<<<2048, 256, 0, stream>>>(q,  Xq16, (int)NX);
  cvt_f32_f16<<<2048, 256, 0, stream>>>(k,  Xk16, (int)NX);
  cvt_f32_f16<<<2048, 256, 0, stream>>>(v,  Xv16, (int)NX);
  cvt_f32_f16<<<1024, 256, 0, stream>>>(Wq, Wq16, (int)NW);
  cvt_f32_f16<<<1024, 256, 0, stream>>>(Wk, Wk16, (int)NW);
  cvt_f32_f16<<<1024, 256, 0, stream>>>(Wv, Wv16, (int)NW);
  cvt_f32_f16<<<1024, 256, 0, stream>>>(Wo, Wo16, (int)NW);

  const dim3 pg(EMBED / 128, (BSZ * SEQ) / 128);   // 8 x 32 blocks
  // Q projection: fold 1/sqrt(HDIM) = 0.125 into Q.
  proj_gemm<0><<<pg, 256, 0, stream>>>(Xq16, Wq16, bq, Q16, nullptr, 0.125f);
  proj_gemm<0><<<pg, 256, 0, stream>>>(Xk16, Wk16, bk, K16, nullptr, 1.0f);
  proj_gemm<0><<<pg, 256, 0, stream>>>(Xv16, Wv16, bv, V16, nullptr, 1.0f);

  flash_attn<<<dim3(SEQ / 128, BSZ * HEADS), 256, 0, stream>>>(Q16, K16, V16, O16);

  proj_gemm<1><<<pg, 256, 0, stream>>>(O16, Wo16, bo, nullptr, (float*)d_out, 1.0f);
}
